// BlockMoVaE_84241488544008
// MI455X (gfx1250) — matrix operand order repaired and checked by reading, not compiled, not run on hardware
//
#include <hip/hip_runtime.h>
#include <hip/hip_bf16.h>

#define USE_ASYNC_COPY 1
#define USE_TDM 1

#define B_    2
#define T_    2048
#define C_    1024
#define NH_   16
#define NKV_  8
#define HD_   64
#define N_TOK (B_*T_)
#define E_MLP_ 8
#define E_VE_  2
#define E_TOT  10
#define HID_  1024
#define VOCAB_ 32000

typedef unsigned short u16;
typedef __bf16 v8bf  __attribute__((ext_vector_type(8)));
typedef __bf16 v16bf __attribute__((ext_vector_type(16)));
typedef float  v8f   __attribute__((ext_vector_type(8)));
typedef __attribute__((address_space(3))) char lds_char;

union FragU { v16bf v; v8bf h[2]; };

__device__ __forceinline__ u16 f2bf(float f){
  unsigned u = __float_as_uint(f);
  u += 0x7fffu + ((u >> 16) & 1u);            // round-to-nearest-even
  return (u16)(u >> 16);
}
__device__ __forceinline__ float bf2f(u16 b){ return __uint_as_float(((unsigned)b) << 16); }

__device__ __forceinline__ v8f wmma_bf16(v16bf a, v16bf b, v8f c){
  return __builtin_amdgcn_wmma_f32_16x16x32_bf16(false, a, false, b, (short)0, c, false, false);
}

__device__ __forceinline__ unsigned lds_off(const void* p){
  return (unsigned)(unsigned long long)(lds_char*)p;   // addrspace(3) offset
}

// ---- cross-lane xor within 16-lane groups via v_permlane16_b32 (pure VALU) ----
__device__ __forceinline__ float pl16(float v, unsigned sl, unsigned sh){
  int r = __builtin_amdgcn_permlane16(__float_as_int(v), __float_as_int(v),
                                      (int)sl, (int)sh, false, false);
  return __int_as_float(r);
}
__device__ __forceinline__ float redmax16(float v){
  v = fmaxf(v, pl16(v, 0x67452301u, 0xEFCDAB89u));   // xor 1
  v = fmaxf(v, pl16(v, 0x54761032u, 0xDCFE98BAu));   // xor 2
  v = fmaxf(v, pl16(v, 0x32107654u, 0xBA98FEDCu));   // xor 4
  v = fmaxf(v, pl16(v, 0xFEDCBA98u, 0x76543210u));   // xor 8
  return v;
}
__device__ __forceinline__ float redsum16(float v){
  v += pl16(v, 0x67452301u, 0xEFCDAB89u);
  v += pl16(v, 0x54761032u, 0xDCFE98BAu);
  v += pl16(v, 0x32107654u, 0xBA98FEDCu);
  v += pl16(v, 0xFEDCBA98u, 0x76543210u);
  return v;
}

// ---- CDNA5 async global->LDS copy (ASYNCcnt) ----
__device__ __forceinline__ void async_copy_b128(void* lds_ptr, const void* gptr){
#if USE_ASYNC_COPY
  unsigned loff = lds_off(lds_ptr);
  asm volatile("global_load_async_to_lds_b128 %0, %1, off"
               :: "v"(loff), "v"(gptr) : "memory");
#else
  *(v8bf*)lds_ptr = *(const v8bf*)gptr;
#endif
}
__device__ __forceinline__ void wait_async0(){
#if USE_ASYNC_COPY
  asm volatile("s_wait_asynccnt 0x0" ::: "memory");
#endif
}

// ---- CDNA5 Tensor Data Mover: 2D weight tile (128 rows x 64 bf16) -> padded LDS ----
#if USE_TDM
typedef unsigned int u32x4 __attribute__((ext_vector_type(4)));
typedef int          i32x8 __attribute__((ext_vector_type(8)));
typedef int          i32x4 __attribute__((ext_vector_type(4)));

__device__ __forceinline__ void tdm_load_tile_b(const u16* gsrc, unsigned lds_addr, int K){
  unsigned long long ga = (unsigned long long)gsrc;
  u32x4 g0;
  g0[0] = 1u;                                              // count=1 (user descriptor)
  g0[1] = lds_addr;                                        // LDS byte address
  g0[2] = (unsigned)(ga & 0xffffffffull);                  // global_addr[31:0]
  g0[3] = (unsigned)((ga >> 32) & 0x01ffffffu) | (2u << 30); // addr[56:32] | type=2 (image)
  i32x8 g1;
  g1[0] = (1 << 16)      // data_size = 2 bytes
        | (1 << 20)      // pad_enable: pad LDS rows
        | (4 << 22)      // pad_interval: 2^(4+1)=32 DWORDs (one 128B row)
        | (3 << 25);     // pad_amount: 3+1=4 DWORDs (16B) -> row pitch 144B = APAD
  g1[1] = (64 & 0xffff) << 16;   // tensor_dim0 = 64 (bits 79:48, low half)
  g1[2] = (128 << 16);           // tensor_dim0 hi = 0 | tensor_dim1 = 128 (bits 95:80)
  g1[3] = (64 << 16);            // tensor_dim1 hi = 0 | tile_dim0 = 64 (bits 127:112)
  g1[4] = 128;                   // tile_dim1 = 128 (bits 143:128), tile_dim2 = 0
  g1[5] = K;                     // tensor_dim0_stride[31:0] (elements)
  g1[6] = 0;                     // stride hi, tensor_dim1_stride lo
  g1[7] = 0;
  i32x4 z4 = {0, 0, 0, 0};       // groups 2/3: tile_dim3/4 = 0 -> unused (2D)
  i32x8 z8 = {0, 0, 0, 0, 0, 0, 0, 0};
  // this toolchain's 6-arg form: (g0, g1, g2, g3, extra_g, cpol)
  __builtin_amdgcn_tensor_load_to_lds(g0, g1, z4, z4, z8, 0);
}
#endif

// A fragment (16x32, MxK): lane holds row rbase+(lane&15);
// elems 0..7 -> K = k0+kb+j, elems 8..15 -> K = k0+16+kb+(j-8), kb = (lane&16)?8:0
__device__ __forceinline__ v16bf load_fragA(const u16* base, int stride, int rbase, int lane, int k0){
  int r  = rbase + (lane & 15);
  int kb = (lane & 16) ? 8 : 0;
  const u16* p = base + (long)r * stride + (k0 + kb);
  FragU f;
  f.h[0] = *(const v8bf*)(p);
  f.h[1] = *(const v8bf*)(p + 16);
  return f.v;
}

// B fragment (32x16, KxN) for out = A @ W^T with W[N,K] row-major:
// lane holds col n = nbase+(lane&15); elems 0..7 -> K = k0+kb+j, elems 8..15 -> K = k0+16+kb+(j-8), kb = (lane&16)?8:0
__device__ __forceinline__ v16bf load_fragB(const u16* base, int stride, int nbase, int lane, int k0){
  int n = nbase + (lane & 15);
  const u16* p = base + (long)n * stride + (k0 + ((lane & 16) ? 8 : 0));
  FragU f;
  f.h[0] = *(const v8bf*)(p);
  f.h[1] = *(const v8bf*)(p + 16);
  return f.v;
}

// ---------------- f32 -> bf16 convert ----------------
__global__ __launch_bounds__(256)
void k_f32_to_bf16(const float* __restrict__ src, u16* __restrict__ dst, long n){
  long i = (long)blockIdx.x * blockDim.x + threadIdx.x;
  long stride = (long)gridDim.x * blockDim.x;
  for (; i < n; i += stride) dst[i] = f2bf(src[i]);
}

// ---------------- zero ints ----------------
__global__ void k_zero_int(int* __restrict__ p, int n){
  int i = threadIdx.x;
  if (i < n) p[i] = 0;
}

// ---------------- RMSNorm (row of C) -> bf16 ----------------
__global__ __launch_bounds__(256)
void k_rmsnorm_bf16(const float* __restrict__ x, u16* __restrict__ out, int C){
  int n = blockIdx.x, tid = threadIdx.x;
  const float* row = x + (long)n * C;
  float ss = 0.f;
  for (int c = tid; c < C; c += 256){ float v = row[c]; ss += v*v; }
  __shared__ float red[256];
  red[tid] = ss; __syncthreads();
  for (int st = 128; st > 0; st >>= 1){ if (tid < st) red[tid] += red[tid + st]; __syncthreads(); }
  float s = rsqrtf(red[0] / (float)C + 1e-6f);
  for (int c = tid; c < C; c += 256) out[(long)n * C + c] = f2bf(row[c] * s);
}

// ---------------- RoPE + head-RMS + pack to [B][H][T][64] bf16 ----------------
__global__ __launch_bounds__(256)
void k_pack_rope(const float* __restrict__ src, const float* __restrict__ cosT,
                 const float* __restrict__ sinT, u16* __restrict__ dst,
                 int H, float scale, int total){
  int idx = blockIdx.x * blockDim.x + threadIdx.x;
  if (idx >= total) return;
  int h = idx % H, n = idx / H;
  int t = n % T_, b = n / T_;
  const float* s  = src  + (long)n * (H * HD_) + h * HD_;
  const float* cp = cosT + (long)t * 32;
  const float* sp = sinT + (long)t * 32;
  float r[64];
  #pragma unroll
  for (int j = 0; j < 32; j++){
    float a = s[j], b2 = s[j + 32], c = cp[j], si = sp[j];
    r[j]      =  a * c + b2 * si;
    r[j + 32] = -a * si + b2 * c;
  }
  float ss = 0.f;
  #pragma unroll
  for (int j = 0; j < 64; j++) ss += r[j] * r[j];
  float sc = rsqrtf(ss / 64.0f + 1e-6f) * scale;
  u16* d = dst + (((long)(b * H + h)) * T_ + t) * HD_;
  #pragma unroll
  for (int j = 0; j < 64; j++) d[j] = f2bf(r[j] * sc);
}

// ---------------- V pack transposed: [B][NKV][64][T] bf16 ----------------
__global__ __launch_bounds__(256)
void k_pack_vT(const float* __restrict__ vf, u16* __restrict__ vT, int total){
  int idx = blockIdx.x * blockDim.x + threadIdx.x;
  if (idx >= total) return;
  int kvh = idx % NKV_, n = idx / NKV_;
  int t = n % T_, b = n / T_;
  const float* src = vf + (long)n * (NKV_ * HD_) + kvh * HD_;
  u16* dst = vT + (((long)(b * NKV_ + kvh)) * HD_) * T_ + t;
  #pragma unroll
  for (int d = 0; d < 64; d++) dst[(long)d * T_] = f2bf(src[d]);
}

// ---------------- Flash attention: 1 wave per 16-query tile ----------------
__global__ __launch_bounds__(32)
void k_attn(const u16* __restrict__ Q, const u16* __restrict__ Kc,
            const u16* __restrict__ Vt, u16* __restrict__ Y,
            const int* __restrict__ winp){
  __shared__ __align__(16) u16 Pl[16 * 32];
  int lane = threadIdx.x & 31;
  int qt = blockIdx.x, h = blockIdx.y, b = blockIdx.z;
  int m0 = qt * 16;
  int kvh = h >> 1;                       // rep = NH/NKV = 2
  int win = winp[0];
  const u16* qb  = Q  + (((long)(b * NH_  + h  )) * T_) * HD_ + (long)m0 * HD_;
  const u16* kbp = Kc + (((long)(b * NKV_ + kvh)) * T_) * HD_;
  const u16* vbp = Vt + (((long)(b * NKV_ + kvh)) * HD_) * T_;
  v16bf qf0 = load_fragA(qb, HD_, 0, lane, 0);
  v16bf qf1 = load_fragA(qb, HD_, 0, lane, 32);
  v8f o[4] = {};
  float mrow[8], lrow[8];
  #pragma unroll
  for (int r = 0; r < 8; r++){ mrow[r] = -3.0e38f; lrow[r] = 0.0f; }
  int lc = lane & 15;
  int rhi = (lane & 16) ? 8 : 0;
  for (int n0 = 0; n0 < m0 + 16; n0 += 32){
    v8f s0 = {}, s1 = {};
    s0 = wmma_bf16(qf0, load_fragB(kbp, HD_, n0,      lane, 0 ), s0);
    s0 = wmma_bf16(qf1, load_fragB(kbp, HD_, n0,      lane, 32), s0);
    s1 = wmma_bf16(qf0, load_fragB(kbp, HD_, n0 + 16, lane, 0 ), s1);
    s1 = wmma_bf16(qf1, load_fragB(kbp, HD_, n0 + 16, lane, 32), s1);
    // uniform: does this key tile need any masking at all?
    bool needmask = (n0 + 31 > m0) || ((m0 + 15 - n0) > win);
    int colA = n0 + lc, colB = colA + 16;
    #pragma unroll
    for (int r = 0; r < 8; r++){
      int grow = m0 + r + rhi;
      float a = s0[r];
      float c = s1[r];
      if (needmask){
        if (colA > grow || grow - colA > win) a = -3.0e38f;
        if (colB > grow || grow - colB > win) c = -3.0e38f;
      }
      float tm = redmax16(fmaxf(a, c));
      float mnew  = fmaxf(mrow[r], tm);
      float alpha = __expf(mrow[r] - mnew);
      float p0 = __expf(a - mnew);
      float p1 = __expf(c - mnew);
      float sum = redsum16(p0 + p1);
      lrow[r] = lrow[r] * alpha + sum;
      mrow[r] = mnew;
      #pragma unroll
      for (int dt = 0; dt < 4; dt++) o[dt][r] *= alpha;
      int rl = r + rhi;
      Pl[rl * 32 + lc]      = f2bf(p0);
      Pl[rl * 32 + lc + 16] = f2bf(p1);
    }
    v16bf pf = load_fragA(Pl, 32, 0, lane, 0);   // same-wave LDS RAW: DS ops in-order
    #pragma unroll
    for (int dt = 0; dt < 4; dt++){
      v16bf vf = load_fragB(vbp, T_, dt * 16, lane, n0);
      o[dt] = wmma_bf16(pf, vf, o[dt]);
    }
  }
  float linv[8];
  #pragma unroll
  for (int r = 0; r < 8; r++) linv[r] = 1.0f / lrow[r];
  #pragma unroll
  for (int dt = 0; dt < 4; dt++){
    #pragma unroll
    for (int r = 0; r < 8; r++){
      int grow = m0 + r + rhi;
      int d = dt * 16 + lc;
      Y[((long)(b * T_ + grow)) * C_ + h * HD_ + d] = f2bf(o[dt][r] * linv[r]);
    }
  }
}

// ---------------- bf16 WMMA GEMM: async-LDS A staging + TDM B staging ----------------
// out = A[M,K] @ W[N,K]^T ; block tile 128x128, K-step 64, 8 waves (4M x 2N), wave 32x64
// MODE 0: outF = acc
// MODE 1: outF = base + acc
// MODE 4: MoE fc  — A rows gathered via gl[slot] (clamped by *cntp); outB = bf16(relu(acc)^2)
// MODE 5: MoE proj — compacted A; scatter outF[gl[slot]*N+gc] += wgt[token*wstride+widx]*acc
#define KSTEP 64
#define APAD  72   // padded LDS row stride (elems): 144B -> b128 frag reads hit banks {0,4,..,60}

template<int MODE>
__global__ __launch_bounds__(256)
void k_gemm_bf16(const u16* __restrict__ A, const u16* __restrict__ W,
                 int M, int N, int K,
                 float* __restrict__ outF, u16* __restrict__ outB,
                 const float* __restrict__ base,
                 const float* __restrict__ wgt, int wstride, int widx,
                 const int* __restrict__ gl, const int* __restrict__ cntp){
  __shared__ __align__(16) u16 As[2][128 * APAD];
  __shared__ __align__(16) u16 Bs[2][128 * APAD];
  int tid = threadIdx.x;
  int lane = tid & 31, wave = tid >> 5;
  int wm = wave & 3, wn = wave >> 2;          // 4 (M) x 2 (N) waves
  int mBase = blockIdx.y * 128;
  int nBase = blockIdx.x * 128;

  int cnt = 0;
  if (MODE == 4 || MODE == 5){
    cnt = *cntp;
    if (mBase >= cnt) return;                 // compacted-M early exit (uniform)
  }

  // A cooperative copy: 128 rows x 64 cols bf16 = 1024 x 16B chunks; 4 chunks/thread
  const u16* aSrc[4]; int lOff[4];
  #pragma unroll
  for (int i = 0; i < 4; i++){
    int c = tid + 256 * i;
    int rr = c >> 3, ff = (c & 7) * 8;
    lOff[i] = rr * APAD + ff;
    int arow;
    if (MODE == 4){
      int slot = mBase + rr;
      arow = (slot < cnt) ? gl[slot] : 0;     // clamp: poisoned list tail never dereferenced OOB
    } else {
      arow = mBase + rr;
    }
    aSrc[i] = A + (long)arow * K + ff;
  }
#if !USE_TDM
  const u16* bSrc[4];
  #pragma unroll
  for (int i = 0; i < 4; i++){
    int c = tid + 256 * i;
    int rr = c >> 3, ff = (c & 7) * 8;
    bSrc[i] = W + (long)(nBase + rr) * K + ff;
  }
#endif
  auto issue = [&](int buf, int k0){
    #pragma unroll
    for (int i = 0; i < 4; i++)
      async_copy_b128(&As[buf][lOff[i]], aSrc[i] + k0);
#if USE_TDM
    if (wave == 0)
      tdm_load_tile_b(W + (long)nBase * K + k0, lds_off(&Bs[buf][0]), K);
#else
    #pragma unroll
    for (int i = 0; i < 4; i++)
      async_copy_b128(&Bs[buf][lOff[i]], bSrc[i] + k0);
#endif
  };

  v8f acc[2][4] = {};
  int KT = K / KSTEP;
  int mL = wm * 32, nL = wn * 64;
  issue(0, 0);
  for (int kt = 0; kt < KT; kt++){
    wait_async0();                         // my wave's async A copies done
#if USE_TDM
    if (wave == 0) __builtin_amdgcn_s_wait_tensorcnt(0);  // TDM tile done
#endif
    __syncthreads();                       // tile kt resident; all waves done with tile kt-1
    if (kt + 1 < KT) issue((kt + 1) & 1, (kt + 1) * KSTEP);
    const u16* At = As[kt & 1];
    const u16* Bt = Bs[kt & 1];
    #pragma unroll
    for (int ks = 0; ks < KSTEP; ks += 32){
      v16bf a0 = load_fragA(At, APAD, mL,      lane, ks);
      v16bf a1 = load_fragA(At, APAD, mL + 16, lane, ks);
      v16bf b0 = load_fragB(Bt, APAD, nL,      lane, ks);
      v16bf b1 = load_fragB(Bt, APAD, nL + 16, lane, ks);
      v16bf b2 = load_fragB(Bt, APAD, nL + 32, lane, ks);
      v16bf b3 = load_fragB(Bt, APAD, nL + 48, lane, ks);
      acc[0][0] = wmma_bf16(a0, b0, acc[0][0]);
      acc[0][1] = wmma_bf16(a0, b1, acc[0][1]);
      acc[0][2] = wmma_bf16(a0, b2, acc[0][2]);
      acc[0][3] = wmma_bf16(a0, b3, acc[0][3]);
      acc[1][0] = wmma_bf16(a1, b0, acc[1][0]);
      acc[1][1] = wmma_bf16(a1, b1, acc[1][1]);
      acc[1][2] = wmma_bf16(a1, b2, acc[1][2]);
      acc[1][3] = wmma_bf16(a1, b3, acc[1][3]);
    }
  }
  int lc = lane & 15, rhi = (lane & 16) ? 8 : 0;
  #pragma unroll
  for (int i = 0; i < 2; i++)
  #pragma unroll
  for (int j = 0; j < 4; j++)
  #pragma unroll
  for (int r = 0; r < 8; r++){
    long gr = mBase + mL + i * 16 + r + rhi;
    long gc = nBase + nL + j * 16 + lc;
    float val = acc[i][j][r];
    if (MODE == 0) outF[gr * (long)N + gc] = val;
    else if (MODE == 1) outF[gr * (long)N + gc] = base[gr * (long)N + gc] + val;
    else if (MODE == 4){ float t = val > 0.f ? val : 0.f; outB[gr * (long)N + gc] = f2bf(t * t); }
    else if (MODE == 5){
      if (gr < cnt){
        long tokn = gl[gr];
        float w = wgt[tokn * (long)wstride + widx];
        outF[tokn * (long)N + gc] += w * val;
      }
    }
  }
}

// ---------------- router: logits, softmax(10), top-2; build per-expert token lists ----------------
__global__ __launch_bounds__(256)
void k_router(const u16* __restrict__ xf, const float* __restrict__ rw,
              float* __restrict__ rout, float* __restrict__ sparse,
              int* __restrict__ cnt, int* __restrict__ lists){
  int n = blockIdx.x, tid = threadIdx.x;
  __shared__ float red[256];
  __shared__ float logits[E_TOT];
  const u16* xrow = xf + (long)n * C_;
  for (int e = 0; e < E_TOT; e++){
    float s = 0.f;
    const float* wrow = rw + (long)e * C_;
    for (int c = tid; c < C_; c += 256) s += bf2f(xrow[c]) * wrow[c];
    red[tid] = s; __syncthreads();
    for (int st = 128; st > 0; st >>= 1){ if (tid < st) red[tid] += red[tid + st]; __syncthreads(); }
    if (tid == 0) logits[e] = red[0];
    __syncthreads();
  }
  if (tid == 0){
    float mx = -3.0e38f;
    for (int e = 0; e < E_TOT; e++) mx = fmaxf(mx, logits[e]);
    float sum = 0.f, rv[E_TOT];
    for (int e = 0; e < E_TOT; e++){ rv[e] = __expf(logits[e] - mx); sum += rv[e]; }
    float inv = 1.0f / sum;
    for (int e = 0; e < E_TOT; e++){ rv[e] *= inv; rout[(long)n * E_TOT + e] = rv[e]; }
    int i1 = 0;
    for (int e = 1; e < E_TOT; e++) if (rv[e] > rv[i1]) i1 = e;
    int i2 = (i1 == 0) ? 1 : 0;
    for (int e = 0; e < E_TOT; e++) if (e != i1 && rv[e] > rv[i2]) i2 = e;
    float nrm = rv[i1] + rv[i2] + 1e-10f;
    for (int e = 0; e < E_TOT; e++) sparse[(long)n * E_TOT + e] = 0.f;
    sparse[(long)n * E_TOT + i1] = rv[i1] / nrm;
    sparse[(long)n * E_TOT + i2] = rv[i2] / nrm;
    if (i1 < E_MLP_){ int s = atomicAdd(&cnt[i1], 1); lists[i1 * N_TOK + s] = n; }
    if (i2 < E_MLP_){ int s = atomicAdd(&cnt[i2], 1); lists[i2 * N_TOK + s] = n; }
  }
}

// ---------------- value-embedding experts accumulate ----------------
__global__ __launch_bounds__(256)
void k_ve(const float* __restrict__ vet, const int* __restrict__ tok,
          const float* __restrict__ sparse, float* __restrict__ out){
  long i = (long)blockIdx.x * blockDim.x + threadIdx.x;
  if (i >= (long)N_TOK * C_) return;
  long n = i >> 10;          // C_ == 1024
  long c = i & 1023;
  float w0 = sparse[n * E_TOT + E_MLP_ + 0];
  float w1 = sparse[n * E_TOT + E_MLP_ + 1];
  if (w0 == 0.f && w1 == 0.f) return;
  long t = (long)tok[n];
  float acc = 0.f;
  if (w0 != 0.f) acc += w0 * vet[t * C_ + c];
  if (w1 != 0.f) acc += w1 * vet[(long)VOCAB_ * C_ + t * C_ + c];
  out[i] += acc;
}

extern "C" void kernel_launch(void* const* d_in, const int* in_sizes, int n_in,
                              void* d_out, int out_size, void* d_ws, size_t ws_size,
                              hipStream_t stream){
  (void)in_sizes; (void)n_in; (void)out_size; (void)ws_size;
  const float* x    = (const float*)d_in[0];
  const float* cosT = (const float*)d_in[1];
  const float* sinT = (const float*)d_in[2];
  const int*   tok  = (const int*)  d_in[3];
  const int*   win  = (const int*)  d_in[4];
  const float* wq   = (const float*)d_in[5];
  const float* wk   = (const float*)d_in[6];
  const float* wv   = (const float*)d_in[7];
  const float* wo   = (const float*)d_in[8];
  const float* rw   = (const float*)d_in[9];
  const float* fcw  = (const float*)d_in[10];
  const float* pjw  = (const float*)d_in[11];
  const float* vet  = (const float*)d_in[12];

  char* wsb = (char*)d_ws;
  size_t off = 0;
  auto alloc = [&](size_t bytes)->char*{
    char* p = wsb + off; off += (bytes + 255) & ~(size_t)255; return p;
  };
  u16*   wq_b = (u16*)  alloc((size_t)(NH_ * HD_) * C_ * 2);
  u16*   wk_b = (u16*)  alloc((size_t)(NKV_ * HD_) * C_ * 2);
  u16*   wv_b = (u16*)  alloc((size_t)(NKV_ * HD_) * C_ * 2);
  u16*   wo_b = (u16*)  alloc((size_t)C_ * C_ * 2);
  u16*   fc_b = (u16*)  alloc((size_t)E_MLP_ * HID_ * C_ * 2);
  u16*   pj_b = (u16*)  alloc((size_t)E_MLP_ * C_ * HID_ * 2);
  u16*   h_b  = (u16*)  alloc((size_t)N_TOK * C_ * 2);          // rms(x); later rms(x2)
  float* q_f  = (float*)alloc((size_t)N_TOK * C_ * 4);          // reused as hidden bf16
  float* k_f  = (float*)alloc((size_t)N_TOK * NKV_ * HD_ * 4);  // reused as Y bf16
  float* v_f  = (float*)alloc((size_t)N_TOK * NKV_ * HD_ * 4);
  u16*   q_b  = (u16*)  alloc((size_t)N_TOK * C_ * 2);
  u16*   k_b  = (u16*)  alloc((size_t)N_TOK * NKV_ * HD_ * 2);
  u16*   vT_b = (u16*)  alloc((size_t)N_TOK * NKV_ * HD_ * 2);
  float* sparse = (float*)alloc((size_t)N_TOK * E_TOT * 4);
  int*   cnt    = (int*)  alloc((size_t)E_MLP_ * 4);
  int*   lists  = (int*)  alloc((size_t)E_MLP_ * N_TOK * 4);
  u16* hid_b = (u16*)q_f;
  u16* y_b   = (u16*)k_f;
  float* out  = (float*)d_out;
  float* rout = out + (size_t)N_TOK * C_;

  // 1) weight conversion to bf16
  k_f32_to_bf16<<<2048, 256, 0, stream>>>(wq,  wq_b, (long)(NH_ * HD_) * C_);
  k_f32_to_bf16<<<2048, 256, 0, stream>>>(wk,  wk_b, (long)(NKV_ * HD_) * C_);
  k_f32_to_bf16<<<2048, 256, 0, stream>>>(wv,  wv_b, (long)(NKV_ * HD_) * C_);
  k_f32_to_bf16<<<2048, 256, 0, stream>>>(wo,  wo_b, (long)C_ * C_);
  k_f32_to_bf16<<<4096, 256, 0, stream>>>(fcw, fc_b, (long)E_MLP_ * HID_ * C_);
  k_f32_to_bf16<<<4096, 256, 0, stream>>>(pjw, pj_b, (long)E_MLP_ * C_ * HID_);
  k_zero_int<<<1, 32, 0, stream>>>(cnt, E_MLP_);

  // 2) h = rms(x) bf16
  k_rmsnorm_bf16<<<N_TOK, 256, 0, stream>>>(x, h_b, C_);

  // 3) q/k/v projections (WMMA)
  k_gemm_bf16<0><<<dim3(1024/128, N_TOK/128), 256, 0, stream>>>(h_b, wq_b, N_TOK, 1024, 1024, q_f, nullptr, nullptr, nullptr, 0, 0, nullptr, nullptr);
  k_gemm_bf16<0><<<dim3( 512/128, N_TOK/128), 256, 0, stream>>>(h_b, wk_b, N_TOK,  512, 1024, k_f, nullptr, nullptr, nullptr, 0, 0, nullptr, nullptr);
  k_gemm_bf16<0><<<dim3( 512/128, N_TOK/128), 256, 0, stream>>>(h_b, wv_b, N_TOK,  512, 1024, v_f, nullptr, nullptr, nullptr, 0, 0, nullptr, nullptr);

  // 4) RoPE + head-RMS packs (attention scale folded into q)
  k_pack_rope<<<(N_TOK * NH_  + 255) / 256, 256, 0, stream>>>(q_f, cosT, sinT, q_b, NH_,  0.125f, N_TOK * NH_);
  k_pack_rope<<<(N_TOK * NKV_ + 255) / 256, 256, 0, stream>>>(k_f, cosT, sinT, k_b, NKV_, 1.0f,   N_TOK * NKV_);
  k_pack_vT  <<<(N_TOK * NKV_ + 255) / 256, 256, 0, stream>>>(v_f, vT_b, N_TOK * NKV_);

  // 5) flash attention (WMMA) -> Y bf16
  k_attn<<<dim3(T_/16, NH_, B_), 32, 0, stream>>>(q_b, k_b, vT_b, y_b, win);

  // 6) x2 = x + Y @ wo^T   -> d_out
  k_gemm_bf16<1><<<dim3(1024/128, N_TOK/128), 256, 0, stream>>>(y_b, wo_b, N_TOK, 1024, 1024, out, nullptr, x, nullptr, 0, 0, nullptr, nullptr);

  // 7) xf = rms(x2) bf16 ; routing (also builds per-expert compacted token lists)
  k_rmsnorm_bf16<<<N_TOK, 256, 0, stream>>>(out, h_b, C_);
  k_router<<<N_TOK, 256, 0, stream>>>(h_b, rw, rout, sparse, cnt, lists);

  // 8) MoE experts on compacted token lists (worst-case grid, count-based early exit;
  //    serialized launches -> race-free, order-deterministic weighted accumulate)
  for (int e = 0; e < E_MLP_; e++){
    k_gemm_bf16<4><<<dim3(1024/128, N_TOK/128), 256, 0, stream>>>(
        h_b, fc_b + (size_t)e * HID_ * C_, N_TOK, HID_, C_,
        nullptr, hid_b, nullptr, nullptr, 0, 0, lists + (size_t)e * N_TOK, cnt + e);
    k_gemm_bf16<5><<<dim3(1024/128, N_TOK/128), 256, 0, stream>>>(
        hid_b, pj_b + (size_t)e * C_ * HID_, N_TOK, C_, HID_,
        out, nullptr, nullptr, sparse, E_TOT, e, lists + (size_t)e * N_TOK, cnt + e);
  }

  // 9) value-embedding experts
  k_ve<<<(N_TOK * C_ + 255) / 256, 256, 0, stream>>>(vet, tok, sparse, out);
}
